// LearntPrototypes_15015205667287
// MI455X (gfx1250) — matrix operand order repaired and checked by reading, not compiled, not run on hardware
//
#include <hip/hip_runtime.h>
#include <stdint.h>

typedef __attribute__((ext_vector_type(16))) __bf16  v16bf;
typedef __attribute__((ext_vector_type(8)))  float   v8f;
typedef __attribute__((ext_vector_type(4)))  int     v4i;

#define N_DIM  65536
#define P_DIM  2048
#define D_DIM  512
#define MT     128      // block tile rows (embeddings)
#define NT     64       // block tile cols (prototypes)
#define KT     32       // k step (bf16 WMMA depth)
#define ROWU32 20       // padded LDS row stride in dwords (16 data + 4 pad -> conflict-free b128)
#define KSTEPS (D_DIM / KT)

#ifndef __has_builtin
#define __has_builtin(x) 0
#endif

#if __has_builtin(__builtin_amdgcn_global_load_async_to_lds_b128)
#define HAVE_ASYNC_LDS 1
#else
#define HAVE_ASYNC_LDS 0
#endif

// 16-byte global->LDS copy; async DMA (ASYNCcnt) when available.
// Builtin signature (from compiler diagnostic): (v4i AS1*, v4i AS3*, imm int, imm int)
__device__ __forceinline__ void cp16(uint32_t* ldst, const uint32_t* gsrc) {
#if HAVE_ASYNC_LDS
  __builtin_amdgcn_global_load_async_to_lds_b128(
      (__attribute__((address_space(1))) v4i*)gsrc,
      (__attribute__((address_space(3))) v4i*)ldst, 0, 0);
#else
  *(uint4*)ldst = *(const uint4*)gsrc;
#endif
}

template <int N>
__device__ __forceinline__ void wait_async() {
#if HAVE_ASYNC_LDS
#if __has_builtin(__builtin_amdgcn_s_wait_asynccnt)
  __builtin_amdgcn_s_wait_asynccnt(N);
#else
  asm volatile("s_wait_asynccnt %0" ::"i"(N) : "memory");
#endif
#endif
}

// ---- bf16 hi/lo split helpers (truncating split keeps lo exactly representable) ----
__device__ __forceinline__ uint32_t pack_hi(float a, float b) {
  return (__float_as_uint(a) >> 16) | (__float_as_uint(b) & 0xFFFF0000u);
}
__device__ __forceinline__ float bf16_residual(float a) {
  return a - __uint_as_float(__float_as_uint(a) & 0xFFFF0000u);
}

// ============================================================================
// Pass 1: fused fp32 -> packed bf16(hi/lo) conversion + squared row norms.
// One wave32 per row (D=512). hi/lo rows are 256 dwords (512 bf16).
// ============================================================================
__global__ void convert_norm_kernel(const float* __restrict__ x,
                                    uint32_t* __restrict__ hi,
                                    uint32_t* __restrict__ lo,
                                    float* __restrict__ sq, int rows) {
  int row  = blockIdx.x * 8 + (threadIdx.x >> 5);
  int lane = threadIdx.x & 31;
  if (row >= rows) return;
  const float4* r = (const float4*)(x + (size_t)row * D_DIM);
  float s = 0.f;
#pragma unroll
  for (int i = 0; i < 4; ++i) {
    int j = lane + 32 * i;              // float4 index within row (0..127)
    float4 v = r[j];
    s += v.x * v.x + v.y * v.y + v.z * v.z + v.w * v.w;
    uint2 h, l;
    h.x = pack_hi(v.x, v.y);
    h.y = pack_hi(v.z, v.w);
    l.x = pack_hi(bf16_residual(v.x), bf16_residual(v.y));
    l.y = pack_hi(bf16_residual(v.z), bf16_residual(v.w));
    *(uint2*)&hi[(size_t)row * 256 + j * 2] = h;
    *(uint2*)&lo[(size_t)row * 256 + j * 2] = l;
  }
#pragma unroll
  for (int off = 16; off > 0; off >>= 1) s += __shfl_down(s, off);
  if (lane == 0) sq[row] = s;
}

// Plain row norms (fallback path)
__global__ void row_sqnorm_kernel(const float* __restrict__ x,
                                  float* __restrict__ out, int rows) {
  int row  = blockIdx.x * 8 + (threadIdx.x >> 5);
  int lane = threadIdx.x & 31;
  if (row >= rows) return;
  const float4* r = (const float4*)(x + (size_t)row * D_DIM);
  float s = 0.f;
#pragma unroll
  for (int i = 0; i < 4; ++i) {
    float4 v = r[lane + 32 * i];
    s += v.x * v.x + v.y * v.y + v.z * v.z + v.w * v.w;
  }
#pragma unroll
  for (int off = 16; off > 0; off >>= 1) s += __shfl_down(s, off);
  if (lane == 0) out[row] = s;
}

// ============================================================================
// Pass 2 (fast path): bf16x3 WMMA GEMM with double-buffered async LDS staging.
// ============================================================================
union Frag { uint4 q[2]; v16bf v; };

// Stage one k-step's tiles (6 x 16B per thread -> ASYNCcnt += 6 per wave).
__device__ __forceinline__ void stage_tiles(
    int tid, int m0, int n0, int ku32,
    const uint32_t* __restrict__ aHi, const uint32_t* __restrict__ aLo,
    const uint32_t* __restrict__ bHi, const uint32_t* __restrict__ bLo,
    uint32_t* sAhi, uint32_t* sAlo, uint32_t* sBhi, uint32_t* sBlo) {
#pragma unroll
  for (int i = 0; i < 2; ++i) {
    int idx = tid + 256 * i;            // 0..511 : 128 rows x 4 chunks
    int row = idx >> 2, c = idx & 3;
    size_t g = (size_t)(m0 + row) * 256 + ku32 + c * 4;
    cp16(&sAhi[row * ROWU32 + c * 4], &aHi[g]);
    cp16(&sAlo[row * ROWU32 + c * 4], &aLo[g]);
  }
  {
    int row = tid >> 2, c = tid & 3;    // 64 rows x 4 chunks
    size_t g = (size_t)(n0 + row) * 256 + ku32 + c * 4;
    cp16(&sBhi[row * ROWU32 + c * 4], &bHi[g]);
    cp16(&sBlo[row * ROWU32 + c * 4], &bLo[g]);
  }
}

__global__ void __launch_bounds__(256)
proto_dist_bf16_kernel(const uint32_t* __restrict__ aHi, const uint32_t* __restrict__ aLo,
                       const uint32_t* __restrict__ bHi, const uint32_t* __restrict__ bLo,
                       const float* __restrict__ xsq, const float* __restrict__ psq,
                       float* __restrict__ out) {
  __shared__ __align__(16) uint32_t sAhi[2][MT * ROWU32];
  __shared__ __align__(16) uint32_t sAlo[2][MT * ROWU32];
  __shared__ __align__(16) uint32_t sBhi[2][NT * ROWU32];
  __shared__ __align__(16) uint32_t sBlo[2][NT * ROWU32];

  const int tid  = threadIdx.x;
  const int lane = tid & 31;
  const int wv   = tid >> 5;
  const int wm   = wv >> 1;       // 0..3
  const int wn   = wv & 1;        // 0..1
  const int l16  = lane & 15;
  const int hh   = lane >> 4;

  const int n0 = blockIdx.x * NT;
  const int m0 = blockIdx.y * MT;

  v8f acc[2][2] = {};

  int cur = 0;
  stage_tiles(tid, m0, n0, 0, aHi, aLo, bHi, bLo,
              sAhi[0], sAlo[0], sBhi[0], sBlo[0]);

  for (int s = 0; s < KSTEPS; ++s) {
    if (s + 1 < KSTEPS) {
      // Prefetch next k-step into the other buffer (overlaps with WMMA below).
      stage_tiles(tid, m0, n0, (s + 1) * (KT / 2), aHi, aLo, bHi, bLo,
                  sAhi[cur ^ 1], sAlo[cur ^ 1], sBhi[cur ^ 1], sBlo[cur ^ 1]);
      wait_async<6>();            // in-order completion: previous buffer done
    } else {
      wait_async<0>();
    }
    __syncthreads();

    // ---- fragments from buf[cur] (ISA 7.12.2 lane layouts) ----
    Frag aFHi[2], aFLo[2], bFHi[2], bFLo[2];
#pragma unroll
    for (int mi = 0; mi < 2; ++mi) {
      int r = wm * 32 + mi * 16 + l16;
      const uint32_t* rh = &sAhi[cur][r * ROWU32];
      const uint32_t* rl = &sAlo[cur][r * ROWU32];
      aFHi[mi].q[0] = *(const uint4*)(rh + (hh ? 4 : 0));
      aFHi[mi].q[1] = *(const uint4*)(rh + (hh ? 12 : 8));
      aFLo[mi].q[0] = *(const uint4*)(rl + (hh ? 4 : 0));
      aFLo[mi].q[1] = *(const uint4*)(rl + (hh ? 12 : 8));
    }
#pragma unroll
    for (int ni = 0; ni < 2; ++ni) {
      int c = wn * 32 + ni * 16 + l16;
      const uint32_t* ch = &sBhi[cur][c * ROWU32];
      const uint32_t* cl = &sBlo[cur][c * ROWU32];
      bFHi[ni].q[0] = *(const uint4*)(ch + (hh ? 4 : 0));
      bFHi[ni].q[1] = *(const uint4*)(ch + (hh ? 12 : 8));
      bFLo[ni].q[0] = *(const uint4*)(cl + (hh ? 4 : 0));
      bFLo[ni].q[1] = *(const uint4*)(cl + (hh ? 12 : 8));
    }

    // ---- bf16x3 emulated-fp32 accumulation: hi*hi + lo*hi + hi*lo ----
#pragma unroll
    for (int mi = 0; mi < 2; ++mi)
#pragma unroll
      for (int ni = 0; ni < 2; ++ni) {
        acc[mi][ni] = __builtin_amdgcn_wmma_f32_16x16x32_bf16(
            false, aFHi[mi].v, false, bFHi[ni].v, (short)0, acc[mi][ni], false, false);
        acc[mi][ni] = __builtin_amdgcn_wmma_f32_16x16x32_bf16(
            false, aFLo[mi].v, false, bFHi[ni].v, (short)0, acc[mi][ni], false, false);
        acc[mi][ni] = __builtin_amdgcn_wmma_f32_16x16x32_bf16(
            false, aFHi[mi].v, false, bFLo[ni].v, (short)0, acc[mi][ni], false, false);
      }

    __syncthreads();   // all waves done reading buf[cur] before it is restaged
    cur ^= 1;
  }

  // ---- fused epilogue: -sqrt(max(xsq - 2*cross + psq, 0)) ----
#pragma unroll
  for (int mi = 0; mi < 2; ++mi) {
#pragma unroll
    for (int ni = 0; ni < 2; ++ni) {
      int gmBase = m0 + wm * 32 + mi * 16 + hh * 8;
      int gn     = n0 + wn * 32 + ni * 16 + l16;
      float pq = psq[gn];
#pragma unroll
      for (int j = 0; j < 8; ++j) {
        int gm = gmBase + j;
        float d2 = fmaxf(xsq[gm] - 2.0f * acc[mi][ni][j] + pq, 0.0f);
        out[(size_t)gm * P_DIM + gn] = -__builtin_sqrtf(d2);
      }
    }
  }
}

// ============================================================================
// Fallback: in-kernel fp32->bf16 conversion (used only if ws is too small).
// ============================================================================
__global__ void __launch_bounds__(256)
proto_dist_kernel(const float* __restrict__ emb, const float* __restrict__ proto,
                  const float* __restrict__ xsq, const float* __restrict__ psq,
                  float* __restrict__ out) {
  __shared__ __align__(16) uint32_t sAhi[MT * ROWU32];
  __shared__ __align__(16) uint32_t sAlo[MT * ROWU32];
  __shared__ __align__(16) uint32_t sBhi[NT * ROWU32];
  __shared__ __align__(16) uint32_t sBlo[NT * ROWU32];

  const int tid  = threadIdx.x;
  const int lane = tid & 31;
  const int wv   = tid >> 5;
  const int wm   = wv >> 1;
  const int wn   = wv & 1;
  const int l16  = lane & 15;
  const int hh   = lane >> 4;

  const int n0 = blockIdx.x * NT;
  const int m0 = blockIdx.y * MT;

  v8f acc[2][2] = {};

  for (int k0 = 0; k0 < D_DIM; k0 += KT) {
    __syncthreads();
#pragma unroll
    for (int i = 0; i < 4; ++i) {
      int idx = tid + 256 * i;
      int row = idx >> 3;
      int c   = idx & 7;
      const float4 v = *(const float4*)(emb + (size_t)(m0 + row) * D_DIM + k0 + c * 4);
      uint32_t base = row * ROWU32 + c * 2;
      sAhi[base]     = pack_hi(v.x, v.y);
      sAhi[base + 1] = pack_hi(v.z, v.w);
      sAlo[base]     = pack_hi(bf16_residual(v.x), bf16_residual(v.y));
      sAlo[base + 1] = pack_hi(bf16_residual(v.z), bf16_residual(v.w));
    }
#pragma unroll
    for (int i = 0; i < 2; ++i) {
      int idx = tid + 256 * i;
      int row = idx >> 3;
      int c   = idx & 7;
      const float4 v = *(const float4*)(proto + (size_t)(n0 + row) * D_DIM + k0 + c * 4);
      uint32_t base = row * ROWU32 + c * 2;
      sBhi[base]     = pack_hi(v.x, v.y);
      sBhi[base + 1] = pack_hi(v.z, v.w);
      sBlo[base]     = pack_hi(bf16_residual(v.x), bf16_residual(v.y));
      sBlo[base + 1] = pack_hi(bf16_residual(v.z), bf16_residual(v.w));
    }
    __syncthreads();

    Frag aFHi[2], aFLo[2], bFHi[2], bFLo[2];
#pragma unroll
    for (int mi = 0; mi < 2; ++mi) {
      int r = wm * 32 + mi * 16 + l16;
      const uint32_t* rh = &sAhi[r * ROWU32];
      const uint32_t* rl = &sAlo[r * ROWU32];
      aFHi[mi].q[0] = *(const uint4*)(rh + (hh ? 4 : 0));
      aFHi[mi].q[1] = *(const uint4*)(rh + (hh ? 12 : 8));
      aFLo[mi].q[0] = *(const uint4*)(rl + (hh ? 4 : 0));
      aFLo[mi].q[1] = *(const uint4*)(rl + (hh ? 12 : 8));
    }
#pragma unroll
    for (int ni = 0; ni < 2; ++ni) {
      int c = wn * 32 + ni * 16 + l16;
      const uint32_t* ch = &sBhi[c * ROWU32];
      const uint32_t* cl = &sBlo[c * ROWU32];
      bFHi[ni].q[0] = *(const uint4*)(ch + (hh ? 4 : 0));
      bFHi[ni].q[1] = *(const uint4*)(ch + (hh ? 12 : 8));
      bFLo[ni].q[0] = *(const uint4*)(cl + (hh ? 4 : 0));
      bFLo[ni].q[1] = *(const uint4*)(cl + (hh ? 12 : 8));
    }

#pragma unroll
    for (int mi = 0; mi < 2; ++mi)
#pragma unroll
      for (int ni = 0; ni < 2; ++ni) {
        acc[mi][ni] = __builtin_amdgcn_wmma_f32_16x16x32_bf16(
            false, aFHi[mi].v, false, bFHi[ni].v, (short)0, acc[mi][ni], false, false);
        acc[mi][ni] = __builtin_amdgcn_wmma_f32_16x16x32_bf16(
            false, aFLo[mi].v, false, bFHi[ni].v, (short)0, acc[mi][ni], false, false);
        acc[mi][ni] = __builtin_amdgcn_wmma_f32_16x16x32_bf16(
            false, aFHi[mi].v, false, bFLo[ni].v, (short)0, acc[mi][ni], false, false);
      }
  }

#pragma unroll
  for (int mi = 0; mi < 2; ++mi) {
#pragma unroll
    for (int ni = 0; ni < 2; ++ni) {
      int gmBase = m0 + wm * 32 + mi * 16 + hh * 8;
      int gn     = n0 + wn * 32 + ni * 16 + l16;
      float pq = psq[gn];
#pragma unroll
      for (int j = 0; j < 8; ++j) {
        int gm = gmBase + j;
        float d2 = fmaxf(xsq[gm] - 2.0f * acc[mi][ni][j] + pq, 0.0f);
        out[(size_t)gm * P_DIM + gn] = -__builtin_sqrtf(d2);
      }
    }
  }
}

extern "C" void kernel_launch(void* const* d_in, const int* in_sizes, int n_in,
                              void* d_out, int out_size, void* d_ws, size_t ws_size,
                              hipStream_t stream) {
  const float* emb   = (const float*)d_in[0];   // [65536, 512] fp32
  const float* proto = (const float*)d_in[1];   // [2048, 512]  fp32
  float* out = (float*)d_out;                   // [65536, 2048] fp32

  // Workspace layout (bytes)
  const size_t off_xsq = 0;
  const size_t off_psq = (size_t)N_DIM * 4;                    // 262144
  const size_t off_ahi = off_psq + (size_t)P_DIM * 4 + 2048;   // 272384, 1KB-aligned
  const size_t szA     = (size_t)N_DIM * D_DIM * 2;            // 64 MB
  const size_t szB     = (size_t)P_DIM * D_DIM * 2;            // 2 MB
  const size_t off_alo = off_ahi + szA;
  const size_t off_bhi = off_alo + szA;
  const size_t off_blo = off_bhi + szB;
  const size_t needed  = off_blo + szB;

  uint8_t* ws  = (uint8_t*)d_ws;
  float*   xsq = (float*)(ws + off_xsq);
  float*   psq = (float*)(ws + off_psq);

  dim3 grid(P_DIM / NT, N_DIM / MT);   // x: prototype tiles (32), y: embedding tiles (512)

  if (ws_size >= needed) {
    uint32_t* aHi = (uint32_t*)(ws + off_ahi);
    uint32_t* aLo = (uint32_t*)(ws + off_alo);
    uint32_t* bHi = (uint32_t*)(ws + off_bhi);
    uint32_t* bLo = (uint32_t*)(ws + off_blo);
    convert_norm_kernel<<<N_DIM / 8, 256, 0, stream>>>(emb, aHi, aLo, xsq, N_DIM);
    convert_norm_kernel<<<P_DIM / 8, 256, 0, stream>>>(proto, bHi, bLo, psq, P_DIM);
    proto_dist_bf16_kernel<<<grid, 256, 0, stream>>>(aHi, aLo, bHi, bLo, xsq, psq, out);
  } else {
    row_sqnorm_kernel<<<N_DIM / 8, 256, 0, stream>>>(emb, xsq, N_DIM);
    row_sqnorm_kernel<<<P_DIM / 8, 256, 0, stream>>>(proto, psq, P_DIM);
    proto_dist_kernel<<<grid, 256, 0, stream>>>(emb, proto, xsq, psq, out);
  }
}
